// FlavorGAT_17695265259560
// MI455X (gfx1250) — matrix operand order repaired and checked by reading, not compiled, not run on hardware
//
#include <hip/hip_runtime.h>
#include <hip/hip_bf16.h>

// ---------------------------------------------------------------------------
// FlavorGAT forward for MI455X (gfx1250, wave32, WMMA).
// GEMMs: v_wmma_f32_16x16x32_bf16, operands pre-converted to bf16 once
// (activations) / prepacked into fragment layout (weights). Wave tile 32x64
// (2x4 WMMA), block tile 128x128. K-loop unrolled x2 with ping-pong register
// sets (no rotation moves), constant-stride pointer advance.
// Edge softmax: 3-pass (atomic max, atomic sum, weighted atomic scatter);
// node matrices fit in 192MB L2 so gathers/atomics stay on-chip.
// ---------------------------------------------------------------------------

typedef __attribute__((ext_vector_type(16))) __bf16 v16bf;
typedef __attribute__((ext_vector_type(8)))  __bf16 v8bf;
typedef __attribute__((ext_vector_type(4)))  __bf16 v4bf;
typedef __attribute__((ext_vector_type(8)))  float  v8f;

#define N_ING 20000
#define N_MOL 40000
#define F_ING 128
#define F_MOL 64
#define HID   256
#define EMB   128

// ---------------------------------------------------------------------------
// Conversion / prepack kernels
// ---------------------------------------------------------------------------

// x (f32, n elems, n % 4 == 0) -> y (bf16), 4 elems/thread, hardware cvt
__global__ __launch_bounds__(256) void cvt_bf16_kernel(const float* __restrict__ x,
                                                       __bf16* __restrict__ y, int n) {
  int i = (blockIdx.x * 256 + threadIdx.x) * 4;
  if (i >= n) return;
  float4 v = *(const float4*)(x + i);
  v4bf o;
  o[0] = (__bf16)v.x; o[1] = (__bf16)v.y; o[2] = (__bf16)v.z; o[3] = (__bf16)v.w;
  *(v4bf*)(y + i) = o;
}

// Prepack W[K,N] (f32 row-major) into per-lane-contiguous WMMA B fragments:
// tile t = (k0/32)*(N/16) + (n0/16); lane holds 16 bf16 at Wp[(t*32+lane)*16]:
// element e -> W[k0 + (lane&16 ? 8:0) + (e<8 ? e : e+8)][n0 + (lane&15)]
__global__ __launch_bounds__(256) void prepack_w_kernel(const float* __restrict__ W,
                                                        __bf16* __restrict__ Wp,
                                                        int K, int N) {
  int idx = blockIdx.x * 256 + threadIdx.x;
  int total = (K >> 5) * (N >> 4) * 32;
  if (idx >= total) return;
  int lane = idx & 31;
  int tile = idx >> 5;
  int ntn  = N >> 4;
  int k0   = (tile / ntn) << 5;
  int n    = ((tile % ntn) << 4) + (lane & 15);
  int kb   = k0 + ((lane & 16) ? 8 : 0);
  const float* src = W + (size_t)kb * N + n;
  __bf16* out = Wp + (size_t)idx * 16;
#pragma unroll
  for (int e = 0; e < 16; ++e) out[e] = (__bf16)src[(size_t)(e < 8 ? e : e + 8) * N];
}

// ---------------------------------------------------------------------------
// WMMA GEMM: Y[M,N] = act(X[M,K] @ W[K,N] + bias), X bf16, W prepacked bf16.
// K % 64 == 0 (steps even), N % 128 == 0. Block: 8 waves (4x2), block tile
// 128x128, wave tile 32x64 = 2x4 WMMA, ping-pong double buffering.
// ---------------------------------------------------------------------------

// A fragment 16x32 (ISA 7.12.2): lanes 0-15 K {kb..kb+7, kb+16..kb+23} with
// kb = +8 for lanes 16-31. p points at (row, k0 + kb) within X.
static __device__ __forceinline__ v16bf load_a_frag(const __bf16* __restrict__ p) {
  v8bf lo = *(const v8bf*)(p);        // K = kb..kb+7
  v8bf hi = *(const v8bf*)(p + 16);   // K = kb+16..kb+23
  return __builtin_shufflevector(lo, hi, 0, 1, 2, 3, 4, 5, 6, 7,
                                 8, 9, 10, 11, 12, 13, 14, 15);
}

template <int ACT>
__global__ __launch_bounds__(256) void gemm_kernel(const __bf16* __restrict__ X,
                                                   const __bf16* __restrict__ Wp,
                                                   const float* __restrict__ bias,
                                                   float* __restrict__ Y,
                                                   int M, int K, int N) {
  int lane = threadIdx.x & 31;
  int w    = threadIdx.x >> 5;
  int r0   = blockIdx.x * 128 + (w >> 1) * 32;
  int c0   = blockIdx.y * 128 + (w & 1) * 64;
  if (r0 >= M) return;  // uniform per wave -> EXEC stays all-ones for WMMA
  int ntn = N >> 4;

  // hoisted per-lane pointers; advance by constant strides in the k loop
  int m0 = r0 + (lane & 15);      if (m0 >= M) m0 = M - 1;
  int m1 = r0 + 16 + (lane & 15); if (m1 >= M) m1 = M - 1;
  int kb = (lane & 16) ? 8 : 0;
  const __bf16* pa0 = X + (size_t)m0 * K + kb;
  const __bf16* pa1 = X + (size_t)m1 * K + kb;
  const __bf16* pb  = Wp + ((size_t)(c0 >> 4) * 32 + lane) * 16;
  const size_t bstep = (size_t)ntn * 512;  // one k-step of prepacked tiles

  v8f acc0[4] = {v8f{}, v8f{}, v8f{}, v8f{}};
  v8f acc1[4] = {v8f{}, v8f{}, v8f{}, v8f{}};

  // ping-pong fragment sets
  v16bf xa0, xa1, xb0, xb1, xb2, xb3;
  v16bf ya0, ya1, yb0, yb1, yb2, yb3;

  auto load_set = [&](v16bf& A0, v16bf& A1, v16bf& B0, v16bf& B1, v16bf& B2, v16bf& B3,
                      bool advance) {
    if (advance) { pa0 += 32; pa1 += 32; pb += bstep; }
    A0 = load_a_frag(pa0);
    A1 = load_a_frag(pa1);
    B0 = *(const v16bf*)(pb);
    B1 = *(const v16bf*)(pb + 512);
    B2 = *(const v16bf*)(pb + 1024);
    B3 = *(const v16bf*)(pb + 1536);
  };
  auto mm = [&](v16bf A0, v16bf A1, v16bf B0, v16bf B1, v16bf B2, v16bf B3) {
    acc0[0] = __builtin_amdgcn_wmma_f32_16x16x32_bf16(false, A0, false, B0, (short)0, acc0[0], false, false);
    acc0[1] = __builtin_amdgcn_wmma_f32_16x16x32_bf16(false, A0, false, B1, (short)0, acc0[1], false, false);
    acc0[2] = __builtin_amdgcn_wmma_f32_16x16x32_bf16(false, A0, false, B2, (short)0, acc0[2], false, false);
    acc0[3] = __builtin_amdgcn_wmma_f32_16x16x32_bf16(false, A0, false, B3, (short)0, acc0[3], false, false);
    acc1[0] = __builtin_amdgcn_wmma_f32_16x16x32_bf16(false, A1, false, B0, (short)0, acc1[0], false, false);
    acc1[1] = __builtin_amdgcn_wmma_f32_16x16x32_bf16(false, A1, false, B1, (short)0, acc1[1], false, false);
    acc1[2] = __builtin_amdgcn_wmma_f32_16x16x32_bf16(false, A1, false, B2, (short)0, acc1[2], false, false);
    acc1[3] = __builtin_amdgcn_wmma_f32_16x16x32_bf16(false, A1, false, B3, (short)0, acc1[3], false, false);
  };

  int half = (K >> 5) >> 1;  // K % 64 == 0 -> steps even, half >= 1
  load_set(xa0, xa1, xb0, xb1, xb2, xb3, false);
  for (int s = 0; s < half - 1; ++s) {
    load_set(ya0, ya1, yb0, yb1, yb2, yb3, true);   // prefetch odd step
    mm(xa0, xa1, xb0, xb1, xb2, xb3);
    load_set(xa0, xa1, xb0, xb1, xb2, xb3, true);   // prefetch next even step
    mm(ya0, ya1, yb0, yb1, yb2, yb3);
  }
  load_set(ya0, ya1, yb0, yb1, yb2, yb3, true);     // final odd step
  mm(xa0, xa1, xb0, xb1, xb2, xb3);
  mm(ya0, ya1, yb0, yb1, yb2, yb3);

  // C/D layout: col = lane&15, rows r (+8 for upper half-wave)
  int halfm = (lane & 16) ? 8 : 0;
#pragma unroll
  for (int ri = 0; ri < 2; ++ri) {
#pragma unroll
    for (int ci = 0; ci < 4; ++ci) {
      v8f acc = ri ? acc1[ci] : acc0[ci];
      int row = r0 + ri * 16 + halfm;
      int col = c0 + ci * 16 + (lane & 15);
      float bv = bias ? bias[col] : 0.0f;
#pragma unroll
      for (int r = 0; r < 8; ++r) {
        int m = row + r;
        if (m < M) {
          float v = acc[r] + bv;
          if (ACT) v = v > 0.0f ? v : 0.0f;
          Y[(size_t)m * N + col] = v;
        }
      }
    }
  }
}

// ---------------------------------------------------------------------------
// GAT edge / normalization kernels
// ---------------------------------------------------------------------------

// out[n,h] = sum_d hs[n, h*32+d] * att[h,d]
__global__ __launch_bounds__(256) void logits_kernel(const float* __restrict__ hs,
                                                     const float* __restrict__ att,
                                                     float* __restrict__ out, int n) {
  int i = blockIdx.x * 256 + threadIdx.x;
  if (i >= n * 8) return;
  int node = i >> 3, h = i & 7;
  const float4* hp = (const float4*)(hs + (size_t)node * HID + h * 32);
  const float4* ap = (const float4*)(att + h * 32);
  float acc = 0.0f;
#pragma unroll
  for (int q = 0; q < 8; ++q) {
    float4 a = hp[q], b = ap[q];
    acc += a.x * b.x + a.y * b.y + a.z * b.z + a.w * b.w;
  }
  out[i] = acc;
}

__global__ void fill_kernel(float* __restrict__ p, float v, int n) {
  int i = blockIdx.x * 256 + threadIdx.x;
  if (i < n) p[i] = v;
}

// out[i*256+c] = b1[c] + b2[c]  (HeteroConv 'sum' of the two relation biases)
__global__ void init_bias2_kernel(float* __restrict__ out, const float* __restrict__ b1,
                                  const float* __restrict__ b2, int n) {
  int i = blockIdx.x * 256 + threadIdx.x;
  if (i >= n * HID) return;
  int c = i & (HID - 1);
  out[i] = b1[c] + b2[c];
}

static __device__ __forceinline__ float leaky02(float x) {
  return x > 0.0f ? x : 0.2f * x;
}

// pass 1: m[dst,h] = max(e)
__global__ __launch_bounds__(256) void edge_max_kernel(const int* __restrict__ src,
                                                       const int* __restrict__ dst,
                                                       const float* __restrict__ as,
                                                       const float* __restrict__ ad,
                                                       float* __restrict__ m, int E) {
  int i = blockIdx.x * 256 + threadIdx.x;
  if (i >= E * 8) return;
  int e = i >> 3, h = i & 7;
  int sv = src[e], dv = dst[e];
  float v = leaky02(as[sv * 8 + h] + ad[dv * 8 + h]);
  __hip_atomic_fetch_max(&m[dv * 8 + h], v, __ATOMIC_RELAXED, __HIP_MEMORY_SCOPE_AGENT);
}

// pass 2: s[dst,h] = sum exp(e - m[dst,h])
__global__ __launch_bounds__(256) void edge_sum_kernel(const int* __restrict__ src,
                                                       const int* __restrict__ dst,
                                                       const float* __restrict__ as,
                                                       const float* __restrict__ ad,
                                                       const float* __restrict__ m,
                                                       float* __restrict__ s, int E) {
  int i = blockIdx.x * 256 + threadIdx.x;
  if (i >= E * 8) return;
  int e = i >> 3, h = i & 7;
  int sv = src[e], dv = dst[e];
  float v = leaky02(as[sv * 8 + h] + ad[dv * 8 + h]);
  atomicAdd(&s[dv * 8 + h], __expf(v - m[dv * 8 + h]));
}

// pass 3: out[dst] += hs[src] * alpha   (wave per edge, lane covers 8 floats)
__global__ __launch_bounds__(256) void edge_scatter_kernel(const int* __restrict__ src,
                                                           const int* __restrict__ dst,
                                                           const float* __restrict__ as,
                                                           const float* __restrict__ ad,
                                                           const float* __restrict__ m,
                                                           const float* __restrict__ s,
                                                           const float* __restrict__ hs,
                                                           float* __restrict__ out, int E) {
  int lane = threadIdx.x & 31;
  int e = blockIdx.x * 8 + (threadIdx.x >> 5);
  if (e >= E) return;
  int sv = src[e], dv = dst[e];
  int h = lane >> 2;  // columns lane*8..lane*8+7 all belong to head lane/4
  float x = leaky02(as[sv * 8 + h] + ad[dv * 8 + h]);
  float alpha = __expf(x - m[dv * 8 + h]) / (s[dv * 8 + h] + 1e-16f);
  int c = lane * 8;
  const float4* hp = (const float4*)(hs + (size_t)sv * HID + c);
  float4 v0 = hp[0], v1 = hp[1];
  float* op = out + (size_t)dv * HID + c;
  atomicAdd(op + 0, v0.x * alpha);
  atomicAdd(op + 1, v0.y * alpha);
  atomicAdd(op + 2, v0.z * alpha);
  atomicAdd(op + 3, v0.w * alpha);
  atomicAdd(op + 4, v1.x * alpha);
  atomicAdd(op + 5, v1.y * alpha);
  atomicAdd(op + 6, v1.z * alpha);
  atomicAdd(op + 7, v1.w * alpha);
}

// per-column sum / sumsq partials (thread t == column t, 128 rows per block)
__global__ __launch_bounds__(256) void bn_stats_kernel(const float* __restrict__ x,
                                                       float* __restrict__ stats, int n) {
  int c  = threadIdx.x;
  int r0 = blockIdx.x * 128;
  int r1 = r0 + 128 < n ? r0 + 128 : n;
  float sum = 0.0f, ssq = 0.0f;
  for (int r = r0; r < r1; ++r) {
    float v = x[(size_t)r * HID + c];
    sum += v; ssq += v * v;
  }
  atomicAdd(&stats[c], sum);
  atomicAdd(&stats[HID + c], ssq);
}

__global__ __launch_bounds__(256) void bn_apply_kernel(const float* __restrict__ x,
                                                       const float* __restrict__ stats,
                                                       const float* __restrict__ g,
                                                       const float* __restrict__ b,
                                                       float* __restrict__ y, int n) {
  int i = blockIdx.x * 256 + threadIdx.x;
  if (i >= n * HID) return;
  int c = i & (HID - 1);
  float inv_n = 1.0f / (float)n;
  float mu  = stats[c] * inv_n;
  float var = stats[HID + c] * inv_n - mu * mu;
  float v = (x[i] - mu) * rsqrtf(var + 1e-5f) * g[c] + b[c];
  y[i] = v > 0.0f ? v : 0.0f;
}

// ---------------------------------------------------------------------------
// Host orchestration
// ---------------------------------------------------------------------------

static void cvt_bf16(hipStream_t st, const float* x, __bf16* y, int n) {
  cvt_bf16_kernel<<<(n / 4 + 255) / 256, 256, 0, st>>>(x, y, n);
}

// prepack W then run GEMM (X already bf16)
static void gemm(hipStream_t st, const __bf16* X, const float* W, __bf16* Wp,
                 const float* b, float* Y, int M, int K, int N, bool relu) {
  int total = (K >> 5) * (N >> 4) * 32;
  prepack_w_kernel<<<(total + 255) / 256, 256, 0, st>>>(W, Wp, K, N);
  dim3 grid((M + 127) / 128, N / 128);
  if (relu) gemm_kernel<1><<<grid, 256, 0, st>>>(X, Wp, b, Y, M, K, N);
  else      gemm_kernel<0><<<grid, 256, 0, st>>>(X, Wp, b, Y, M, K, N);
}

static void gat_edges(hipStream_t st, const int* src, const int* dst,
                      const float* as, const float* ad, float* m, float* s,
                      const float* hs, float* out, int E, int n_dst) {
  int nh = n_dst * 8;
  fill_kernel<<<(nh + 255) / 256, 256, 0, st>>>(m, -3.402823466e38f, nh);
  fill_kernel<<<(nh + 255) / 256, 256, 0, st>>>(s, 0.0f, nh);
  int eh = E * 8;
  edge_max_kernel<<<(eh + 255) / 256, 256, 0, st>>>(src, dst, as, ad, m, E);
  edge_sum_kernel<<<(eh + 255) / 256, 256, 0, st>>>(src, dst, as, ad, m, s, E);
  edge_scatter_kernel<<<(E + 7) / 8, 256, 0, st>>>(src, dst, as, ad, m, s, hs, out, E);
}

extern "C" void kernel_launch(void* const* d_in, const int* in_sizes, int n_in,
                              void* d_out, int out_size, void* d_ws, size_t ws_size,
                              hipStream_t stream) {
  (void)n_in; (void)out_size; (void)ws_size;
  auto F = [&](int i) { return (const float*)d_in[i]; };
  auto I = [&](int i) { return (const int*)d_in[i]; };

  const float* x_ing = F(0);
  const float* x_mol = F(1);
  const int *con_src = I(2), *con_dst = I(3);
  const int *rev_src = I(4), *rev_dst = I(5);
  const int *co_src  = I(6), *co_dst  = I(7);
  const int *sim_src = I(8), *sim_dst = I(9);
  int E_con = in_sizes[2], E_rev = in_sizes[4], E_co = in_sizes[6], E_sim = in_sizes[8];

  // params flattened as a JAX pytree: top-level dict keys sorted:
  // embed_ing{W,b}, embed_mol{W,b}, layers[3]{bn_ing{b,g}, bn_mol{b,g},
  //   co{W,att_dst,att_src,bias}, contains{W_dst,W_src,att_dst,att_src,bias},
  //   rev{W_dst,W_src,att_dst,att_src,bias}, sim{W,att_dst,att_src,bias}},
  // proj_ing{W,b}, proj_mol{W,b}
  const float* embed_ing_W = F(10);
  const float* embed_ing_b = F(11);
  const float* embed_mol_W = F(12);
  const float* embed_mol_b = F(13);
  const float* proj_ing_W = F(80);
  const float* proj_ing_b = F(81);
  const float* proj_mol_W = F(82);
  const float* proj_mol_b = F(83);

  // workspace layout (float-granular offsets)
  float* ws = (float*)d_ws;
  float* h_ing   = ws;                                 // N_ING*HID f32
  float* h_mol   = h_ing   + (size_t)N_ING * HID;      // N_MOL*HID f32
  float* new_ing = h_mol   + (size_t)N_MOL * HID;      // N_ING*HID f32
  float* new_mol = new_ing + (size_t)N_ING * HID;      // N_MOL*HID f32
  float* bufA    = new_mol + (size_t)N_MOL * HID;      // N_ING*HID f32 (ing-side feats)
  float* bufB    = bufA    + (size_t)N_ING * HID;      // N_MOL*HID f32 (mol-side feats)
  float* a_s     = bufB    + (size_t)N_MOL * HID;      // N_MOL*8
  float* a_d     = a_s  + (size_t)N_MOL * 8;
  float* mbuf    = a_d  + (size_t)N_MOL * 8;
  float* sbuf    = mbuf + (size_t)N_MOL * 8;
  float* stats   = sbuf + (size_t)N_MOL * 8;           // 2*HID
  float* bf_base = stats + 2 * HID;
  __bf16* h_ing_bf = (__bf16*)bf_base;                          // N_ING*HID bf16
  __bf16* h_mol_bf = h_ing_bf + (size_t)N_ING * HID;            // N_MOL*HID bf16
  __bf16* x_ing_bf = h_mol_bf + (size_t)N_MOL * HID;            // N_ING*F_ING bf16
  __bf16* x_mol_bf = x_ing_bf + (size_t)N_ING * F_ING;          // N_MOL*F_MOL bf16
  __bf16* Wp       = x_mol_bf + (size_t)N_MOL * F_MOL;          // HID*HID bf16

  // input projections (convert x once, then WMMA GEMM + ReLU)
  cvt_bf16(stream, x_ing, x_ing_bf, N_ING * F_ING);
  cvt_bf16(stream, x_mol, x_mol_bf, N_MOL * F_MOL);
  gemm(stream, x_ing_bf, proj_ing_W, Wp, proj_ing_b, h_ing, N_ING, F_ING, HID, true);
  gemm(stream, x_mol_bf, proj_mol_W, Wp, proj_mol_b, h_mol, N_MOL, F_MOL, HID, true);

  for (int l = 0; l < 3; ++l) {
    int lb = 14 + 22 * l;
    const float* bn_ing_b = F(lb + 0);
    const float* bn_ing_g = F(lb + 1);
    const float* bn_mol_b = F(lb + 2);
    const float* bn_mol_g = F(lb + 3);
    const float* co_W = F(lb + 4), *co_att_d = F(lb + 5), *co_att_s = F(lb + 6), *co_bias = F(lb + 7);
    const float* c_Wd = F(lb + 8), *c_Ws = F(lb + 9), *c_att_d = F(lb + 10), *c_att_s = F(lb + 11), *c_bias = F(lb + 12);
    const float* r_Wd = F(lb + 13), *r_Ws = F(lb + 14), *r_att_d = F(lb + 15), *r_att_s = F(lb + 16), *r_bias = F(lb + 17);
    const float* si_W = F(lb + 18), *si_att_d = F(lb + 19), *si_att_s = F(lb + 20), *si_bias = F(lb + 21);

    // convert this layer's activations once; reused by all 6 GEMMs below
    cvt_bf16(stream, h_ing, h_ing_bf, N_ING * HID);
    cvt_bf16(stream, h_mol, h_mol_bf, N_MOL * HID);

    // ---- mol destination: contains (ing->mol) + sim (mol->mol) ----
    init_bias2_kernel<<<(N_MOL * HID + 255) / 256, 256, 0, stream>>>(new_mol, c_bias, si_bias, N_MOL);

    gemm(stream, h_ing_bf, c_Ws, Wp, nullptr, bufA, N_ING, HID, HID, false);   // hs
    gemm(stream, h_mol_bf, c_Wd, Wp, nullptr, bufB, N_MOL, HID, HID, false);   // hd
    logits_kernel<<<(N_ING * 8 + 255) / 256, 256, 0, stream>>>(bufA, c_att_s, a_s, N_ING);
    logits_kernel<<<(N_MOL * 8 + 255) / 256, 256, 0, stream>>>(bufB, c_att_d, a_d, N_MOL);
    gat_edges(stream, con_src, con_dst, a_s, a_d, mbuf, sbuf, bufA, new_mol, E_con, N_MOL);

    gemm(stream, h_mol_bf, si_W, Wp, nullptr, bufB, N_MOL, HID, HID, false);   // homo: hs == hd
    logits_kernel<<<(N_MOL * 8 + 255) / 256, 256, 0, stream>>>(bufB, si_att_s, a_s, N_MOL);
    logits_kernel<<<(N_MOL * 8 + 255) / 256, 256, 0, stream>>>(bufB, si_att_d, a_d, N_MOL);
    gat_edges(stream, sim_src, sim_dst, a_s, a_d, mbuf, sbuf, bufB, new_mol, E_sim, N_MOL);

    // ---- ing destination: rev (mol->ing) + co (ing->ing) ----
    init_bias2_kernel<<<(N_ING * HID + 255) / 256, 256, 0, stream>>>(new_ing, r_bias, co_bias, N_ING);

    gemm(stream, h_mol_bf, r_Ws, Wp, nullptr, bufB, N_MOL, HID, HID, false);   // hs
    gemm(stream, h_ing_bf, r_Wd, Wp, nullptr, bufA, N_ING, HID, HID, false);   // hd
    logits_kernel<<<(N_MOL * 8 + 255) / 256, 256, 0, stream>>>(bufB, r_att_s, a_s, N_MOL);
    logits_kernel<<<(N_ING * 8 + 255) / 256, 256, 0, stream>>>(bufA, r_att_d, a_d, N_ING);
    gat_edges(stream, rev_src, rev_dst, a_s, a_d, mbuf, sbuf, bufB, new_ing, E_rev, N_ING);

    gemm(stream, h_ing_bf, co_W, Wp, nullptr, bufA, N_ING, HID, HID, false);   // homo
    logits_kernel<<<(N_ING * 8 + 255) / 256, 256, 0, stream>>>(bufA, co_att_s, a_s, N_ING);
    logits_kernel<<<(N_ING * 8 + 255) / 256, 256, 0, stream>>>(bufA, co_att_d, a_d, N_ING);
    gat_edges(stream, co_src, co_dst, a_s, a_d, mbuf, sbuf, bufA, new_ing, E_co, N_ING);

    // ---- BatchNorm + ReLU back into h_ing / h_mol ----
    fill_kernel<<<2, 256, 0, stream>>>(stats, 0.0f, 2 * HID);
    bn_stats_kernel<<<(N_ING + 127) / 128, 256, 0, stream>>>(new_ing, stats, N_ING);
    bn_apply_kernel<<<(N_ING * HID + 255) / 256, 256, 0, stream>>>(new_ing, stats, bn_ing_g, bn_ing_b, h_ing, N_ING);

    fill_kernel<<<2, 256, 0, stream>>>(stats, 0.0f, 2 * HID);
    bn_stats_kernel<<<(N_MOL + 127) / 128, 256, 0, stream>>>(new_mol, stats, N_MOL);
    bn_apply_kernel<<<(N_MOL * HID + 255) / 256, 256, 0, stream>>>(new_mol, stats, bn_mol_g, bn_mol_b, h_mol, N_MOL);
  }

  // final embeddings straight into d_out: [emb_ing | emb_mol]
  cvt_bf16(stream, h_ing, h_ing_bf, N_ING * HID);
  cvt_bf16(stream, h_mol, h_mol_bf, N_MOL * HID);
  float* out = (float*)d_out;
  gemm(stream, h_ing_bf, embed_ing_W, Wp, embed_ing_b, out,                       N_ING, HID, EMB, false);
  gemm(stream, h_mol_bf, embed_mol_W, Wp, embed_mol_b, out + (size_t)N_ING * EMB, N_MOL, HID, EMB, false);
}
